// TransformerConv_12738872999965
// MI455X (gfx1250) — compile-verified, matrix operand order checked by reading, not run on hardware
//
#include <hip/hip_runtime.h>

// ---------------------------------------------------------------------------
// Types for CDNA5 WMMA (wave32, 16x16x32 bf16 -> f32)
// ---------------------------------------------------------------------------
typedef __bf16 bf16_t;
typedef bf16_t bf16x16 __attribute__((ext_vector_type(16)));
typedef bf16_t bf16x2  __attribute__((ext_vector_type(2)));
typedef float  f32x8   __attribute__((ext_vector_type(8)));

#define DIM   128     // D == H*C
#define HEADS 8
#define CH    16

// Packed f32 -> bf16 conversion: prefer the native v_cvt_pk_bf16_f32.
#if __has_builtin(__builtin_amdgcn_cvt_pk_bf16_f32)
static __device__ __forceinline__ bf16x2 f2bf2(float lo, float hi) {
    return __builtin_amdgcn_cvt_pk_bf16_f32(lo, hi);
}
#else
static __device__ __forceinline__ bf16x2 f2bf2(float lo, float hi) {
    bf16x2 r; r[0] = (bf16_t)lo; r[1] = (bf16_t)hi; return r;
}
#endif

static __device__ __forceinline__ bf16_t f2bf(float f) {
    return f2bf2(f, f)[0];
}

// Build an A-fragment (16x32 bf16, one K-step) from a row-major fp32 matrix.
// A layout (ISA 7.12.2): lane L holds row M = L&15; half hi = L>>4 selects
// K sub-blocks {8*hi..8*hi+7} and {16+8*hi..16+8*hi+7}.
static __device__ __forceinline__ bf16x16 load_a_frag(const float* rowbase, int kt, int hi) {
    const float* p = rowbase + kt * 32 + 8 * hi;
    float4 s0 = *(const float4*)(p);
    float4 s1 = *(const float4*)(p + 4);
    float4 s2 = *(const float4*)(p + 16);
    float4 s3 = *(const float4*)(p + 20);
    union { bf16x16 v; bf16x2 p2[8]; } u;
    u.p2[0] = f2bf2(s0.x, s0.y);
    u.p2[1] = f2bf2(s0.z, s0.w);
    u.p2[2] = f2bf2(s1.x, s1.y);
    u.p2[3] = f2bf2(s1.z, s1.w);
    u.p2[4] = f2bf2(s2.x, s2.y);
    u.p2[5] = f2bf2(s2.z, s2.w);
    u.p2[6] = f2bf2(s3.x, s3.y);
    u.p2[7] = f2bf2(s3.z, s3.w);
    return u.v;
}

static __device__ __forceinline__ f32x8 wmma_bf16(bf16x16 a, bf16x16 b, f32x8 c) {
    return __builtin_amdgcn_wmma_f32_16x16x32_bf16(false, a, false, b, (short)0, c, false, false);
}

// ---------------------------------------------------------------------------
// Kernel 1: convert + swizzle the 5 weight matrices (128x128 each) into the
// WMMA B-fragment layout: frag(kt,nt) -> 32 lanes x 16 contiguous bf16.
// B element for lane L, slot i:  K = kt*32 + 16*(i>>3) + 8*(L>>4) + (i&7),
//                                N = nt*16 + (L&15)
// ---------------------------------------------------------------------------
__global__ void cvt_weights_k(const float* __restrict__ Wq, const float* __restrict__ Wk,
                              const float* __restrict__ Wv, const float* __restrict__ Wskip,
                              const float* __restrict__ We, bf16_t* __restrict__ out) {
    int t = blockIdx.x * blockDim.x + threadIdx.x;
    if (t >= 5 * 16384) return;
    int w    = t >> 14;
    int r    = t & 16383;
    int frag = r >> 9;             // 0..31  (kt*8 + nt)
    int kt   = frag >> 3;
    int nt   = frag & 7;
    int li   = r & 511;
    int lane = li >> 4;
    int i    = li & 15;
    int K    = kt * 32 + 16 * (i >> 3) + 8 * (lane >> 4) + (i & 7);
    int Ncol = nt * 16 + (lane & 15);
    const float* W = (w == 0) ? Wq : (w == 1) ? Wk : (w == 2) ? Wv : (w == 3) ? Wskip : We;
    out[t] = f2bf(W[K * DIM + Ncol]);
}

// ---------------------------------------------------------------------------
// Kernel 2: per-call init of segment-max (encoded uint) and denom buffers.
// ---------------------------------------------------------------------------
__global__ void init_nodes_k(unsigned* __restrict__ menc, float* __restrict__ denom, int total) {
    int t = blockIdx.x * blockDim.x + threadIdx.x;
    if (t >= total) return;
    menc[t]  = 0u;     // below encode(-inf)=0x007FFFFF -> acts as -infinity
    denom[t] = 0.0f;
}

// ---------------------------------------------------------------------------
// Kernel 3: node projections.  One wave per 16-node tile.
//   q,k,v -> workspace ;  skip (x@Wskip + bskip) -> d_out (full overwrite)
// ---------------------------------------------------------------------------
__global__ __launch_bounds__(256)
void node_proj_k(const float* __restrict__ x, const bf16_t* __restrict__ wbf,
                 const float* __restrict__ bq, const float* __restrict__ bk,
                 const float* __restrict__ bv, const float* __restrict__ bs,
                 float* __restrict__ q, float* __restrict__ k, float* __restrict__ v,
                 float* __restrict__ out, int ntiles) {
    int wave = (blockIdx.x * blockDim.x + threadIdx.x) >> 5;
    if (wave >= ntiles) return;                       // wave-uniform: EXEC stays all-1
    int lane = threadIdx.x & 31;
    int hi   = lane >> 4;
    int nl   = lane & 15;

    const float* rowbase = x + (size_t)(wave * 16 + nl) * DIM;
    bf16x16 a[4];
#pragma unroll
    for (int kt = 0; kt < 4; ++kt) a[kt] = load_a_frag(rowbase, kt, hi);

#pragma unroll
    for (int w = 0; w < 4; ++w) {
        const bf16_t* W    = wbf + (size_t)w * 16384;
        const float*  bias = (w == 0) ? bq : (w == 1) ? bk : (w == 2) ? bv : bs;
        float*        dst  = (w == 0) ? q  : (w == 1) ? k  : (w == 2) ? v  : out;
#pragma unroll
        for (int nt = 0; nt < 8; ++nt) {
            f32x8 acc = {};
#pragma unroll
            for (int kt = 0; kt < 4; ++kt) {
                bf16x16 b = *(const bf16x16*)(W + (((kt * 8 + nt) * 32 + lane) << 4));
                acc = wmma_bf16(a[kt], b, acc);
            }
            float bval = bias[nt * 16 + nl];
#pragma unroll
            for (int j = 0; j < 8; ++j) {
                int m = 8 * hi + j;   // C/D layout: lane half -> row block
                dst[(size_t)(wave * 16 + m) * DIM + nt * 16 + nl] = acc[j] + bval;
            }
        }
    }
}

// ---------------------------------------------------------------------------
// Kernel 4: edge pass 1.  One wave per 16-edge tile: e = edge_attr@We (WMMA),
// then per head h (== N-tile, since C==16): alpha = q[dst]·(k[src]+e)/4.
// ---------------------------------------------------------------------------
__global__ __launch_bounds__(256)
void edge_pass1_k(const float* __restrict__ ea, const bf16_t* __restrict__ webf,
                  const float* __restrict__ q, const float* __restrict__ k,
                  const long long* __restrict__ ei, float* __restrict__ alpha,
                  long long E, int etiles) {
    int wave = (blockIdx.x * blockDim.x + threadIdx.x) >> 5;
    if (wave >= etiles) return;
    int lane = threadIdx.x & 31;
    int hi   = lane >> 4;
    int nl   = lane & 15;

    const float* rowbase = ea + (size_t)(wave * 16 + nl) * DIM;
    bf16x16 a[4];
#pragma unroll
    for (int kt = 0; kt < 4; ++kt) a[kt] = load_a_frag(rowbase, kt, hi);

    int srcj[8], dstj[8];
#pragma unroll
    for (int j = 0; j < 8; ++j) {
        long long eid = (long long)wave * 16 + 8 * hi + j;
        srcj[j] = (int)ei[eid];
        dstj[j] = (int)ei[E + eid];
    }

#pragma unroll
    for (int h = 0; h < HEADS; ++h) {
        f32x8 acc = {};
#pragma unroll
        for (int kt = 0; kt < 4; ++kt) {
            bf16x16 b = *(const bf16x16*)(webf + (((kt * 8 + h) * 32 + lane) << 4));
            acc = wmma_bf16(a[kt], b, acc);
        }
        float partial[8];
#pragma unroll
        for (int j = 0; j < 8; ++j) {
            float kj = acc[j] + k[(size_t)srcj[j] * DIM + h * CH + nl];
            float qi = q[(size_t)dstj[j] * DIM + h * CH + nl];
            partial[j] = qi * kj;
        }
#pragma unroll
        for (int off = 1; off < 16; off <<= 1)
#pragma unroll
            for (int j = 0; j < 8; ++j)
                partial[j] += __shfl_xor(partial[j], off, 32);
        if (nl == 0) {
#pragma unroll
            for (int j = 0; j < 8; ++j)
                alpha[((size_t)wave * 16 + 8 * hi + j) * HEADS + h] = partial[j] * 0.25f;
        }
    }
}

// ---------------------------------------------------------------------------
// Kernel 5: segment max over (dst, head) using monotonic uint encoding.
// ---------------------------------------------------------------------------
__global__ void seg_max_k(const float* __restrict__ alpha, const long long* __restrict__ ei,
                          unsigned* __restrict__ menc, long long E) {
    long long t = (long long)blockIdx.x * blockDim.x + threadIdx.x;
    if (t >= E * HEADS) return;
    long long e = t >> 3;
    int h = (int)(t & 7);
    int d = (int)ei[E + e];
    unsigned u = __float_as_uint(alpha[t]);
    u = (u >> 31) ? ~u : (u | 0x80000000u);
    atomicMax(&menc[(size_t)d * HEADS + h], u);
}

// ---------------------------------------------------------------------------
// Kernel 6: p = exp(alpha - m[dst]) (stored in place), denom += p.
// ---------------------------------------------------------------------------
__global__ void seg_sum_k(float* __restrict__ alpha, const long long* __restrict__ ei,
                          const unsigned* __restrict__ menc, float* __restrict__ denom,
                          long long E) {
    long long t = (long long)blockIdx.x * blockDim.x + threadIdx.x;
    if (t >= E * HEADS) return;
    long long e = t >> 3;
    int h = (int)(t & 7);
    int d = (int)ei[E + e];
    unsigned u = menc[(size_t)d * HEADS + h];
    float m = (u >> 31) ? __uint_as_float(u & 0x7FFFFFFFu) : __uint_as_float(~u);
    float p = __expf(alpha[t] - m);
    alpha[t] = p;
    atomicAdd(&denom[(size_t)d * HEADS + h], p);
}

// ---------------------------------------------------------------------------
// Kernel 7: edge pass 2.  Recompute e (cheaper than spilling 410 MB), then
// out[dst] += attn * (v[src] + e)  via float atomics.
// ---------------------------------------------------------------------------
__global__ __launch_bounds__(256)
void edge_pass2_k(const float* __restrict__ ea, const bf16_t* __restrict__ webf,
                  const float* __restrict__ v, const float* __restrict__ p,
                  const float* __restrict__ denom, const long long* __restrict__ ei,
                  float* __restrict__ out, long long E, int etiles) {
    int wave = (blockIdx.x * blockDim.x + threadIdx.x) >> 5;
    if (wave >= etiles) return;
    int lane = threadIdx.x & 31;
    int hi   = lane >> 4;
    int nl   = lane & 15;

    const float* rowbase = ea + (size_t)(wave * 16 + nl) * DIM;
    bf16x16 a[4];
#pragma unroll
    for (int kt = 0; kt < 4; ++kt) a[kt] = load_a_frag(rowbase, kt, hi);

    int srcj[8], dstj[8];
#pragma unroll
    for (int j = 0; j < 8; ++j) {
        long long eid = (long long)wave * 16 + 8 * hi + j;
        srcj[j] = (int)ei[eid];
        dstj[j] = (int)ei[E + eid];
    }

#pragma unroll
    for (int h = 0; h < HEADS; ++h) {
        f32x8 acc = {};
#pragma unroll
        for (int kt = 0; kt < 4; ++kt) {
            bf16x16 b = *(const bf16x16*)(webf + (((kt * 8 + h) * 32 + lane) << 4));
            acc = wmma_bf16(a[kt], b, acc);
        }
#pragma unroll
        for (int j = 0; j < 8; ++j) {
            long long eid = (long long)wave * 16 + 8 * hi + j;
            float pj  = p[eid * HEADS + h];
            float dn  = denom[(size_t)dstj[j] * HEADS + h];
            float att = pj / (dn + 1e-16f);
            float vj  = v[(size_t)srcj[j] * DIM + h * CH + nl] + acc[j];
            atomicAdd(&out[(size_t)dstj[j] * DIM + h * CH + nl], att * vj);
        }
    }
}

// ---------------------------------------------------------------------------
// Host launcher
// ---------------------------------------------------------------------------
extern "C" void kernel_launch(void* const* d_in, const int* in_sizes, int n_in,
                              void* d_out, int out_size, void* d_ws, size_t ws_size,
                              hipStream_t stream) {
    const float*     x   = (const float*)d_in[0];
    const float*     ea  = (const float*)d_in[1];
    const float*     Wq  = (const float*)d_in[2];
    const float*     bq  = (const float*)d_in[3];
    const float*     Wk  = (const float*)d_in[4];
    const float*     bk  = (const float*)d_in[5];
    const float*     Wv  = (const float*)d_in[6];
    const float*     bv  = (const float*)d_in[7];
    const float*     We  = (const float*)d_in[8];
    const float*     Wsk = (const float*)d_in[9];
    const float*     bsk = (const float*)d_in[10];
    const long long* ei  = (const long long*)d_in[11];
    float*           out = (float*)d_out;

    const int       N = in_sizes[0] / DIM;      // 50000
    const long long E = in_sizes[1] / DIM;      // 800000

    // workspace layout (all 256B aligned)
    char*     ws    = (char*)d_ws;
    bf16_t*   wbf   = (bf16_t*)ws;                       // 5 * 16384 bf16 = 163840 B
    size_t    off   = 5 * 16384 * sizeof(bf16_t);
    float*    qb    = (float*)(ws + off); off += (size_t)N * DIM * 4;
    float*    kb    = (float*)(ws + off); off += (size_t)N * DIM * 4;
    float*    vb    = (float*)(ws + off); off += (size_t)N * DIM * 4;
    float*    alpha = (float*)(ws + off); off += (size_t)E * HEADS * 4;
    unsigned* menc  = (unsigned*)(ws + off); off += (size_t)N * HEADS * 4;
    float*    denom = (float*)(ws + off); off += (size_t)N * HEADS * 4;

    const int ntiles = N / 16;                 // 3125
    const int etiles = (int)(E / 16);          // 50000

    // 1. weights -> bf16 fragment layout
    cvt_weights_k<<<(5 * 16384 + 255) / 256, 256, 0, stream>>>(Wq, Wk, Wv, Wsk, We, wbf);

    // 2. reset segment buffers (every call; graph replays don't re-poison)
    init_nodes_k<<<(N * HEADS + 255) / 256, 256, 0, stream>>>(menc, denom, N * HEADS);

    // 3. node projections (q,k,v -> ws; skip -> out)
    node_proj_k<<<(ntiles * 32 + 255) / 256, 256, 0, stream>>>(x, wbf, bq, bk, bv, bsk,
                                                               qb, kb, vb, out, ntiles);

    // 4. edge pass 1: alpha
    edge_pass1_k<<<(etiles * 32 + 255) / 256, 256, 0, stream>>>(ea, wbf + 4 * 16384,
                                                                qb, kb, ei, alpha, E, etiles);

    // 5/6. segment softmax statistics
    long long tot = E * HEADS;
    seg_max_k<<<(unsigned)((tot + 255) / 256), 256, 0, stream>>>(alpha, ei, menc, E);
    seg_sum_k<<<(unsigned)((tot + 255) / 256), 256, 0, stream>>>(alpha, ei, menc, denom, E);

    // 7. edge pass 2: aggregate messages into out
    edge_pass2_k<<<(etiles * 32 + 255) / 256, 256, 0, stream>>>(ea, wbf + 4 * 16384,
                                                                vb, alpha, denom, ei, out, E, etiles);
}
